// MXMNet_29996051595847
// MI455X (gfx1250) — matrix operand order repaired and checked by reading, not compiled, not run on hardware
//
#include <hip/hip_runtime.h>
#include <hip/hip_bf16.h>
#include <math.h>

// ---------------------------------------------------------------------------
// MXMNet forward for MI455X (gfx1250, wave32).
// Dense 128-out MLPs -> bf16 WMMA GEMM (v_wmma_f32_16x16x32_bf16) with fused
// gather-concat A assembly and fused bias/SiLU/mul/add epilogue.
// Weights are pre-converted to bf16 (transposed K-major) so the W tile is a
// pure byte copy staged by the Tensor Data Mover (tensor_load_to_lds,
// s_wait_tensorcnt) when available.
// ---------------------------------------------------------------------------

#define DDIM 128
#define TK 32

typedef __bf16 bf16_t;
typedef __attribute__((ext_vector_type(16))) __bf16 bf16x16;
typedef __attribute__((ext_vector_type(8)))  float  f32x8;
typedef __attribute__((ext_vector_type(4)))  unsigned u32x4;
typedef __attribute__((ext_vector_type(8)))  int      i32x8;
typedef __attribute__((ext_vector_type(4)))  int      i32x4;

#if __has_builtin(__builtin_amdgcn_tensor_load_to_lds) && \
    __has_builtin(__builtin_amdgcn_s_wait_tensorcnt)
#define USE_TDM 1
#else
#define USE_TDM 0
#endif

__device__ __forceinline__ float silu_f(float x) { return x / (1.0f + __expf(-x)); }

__device__ __forceinline__ bf16_t to_bf16(float f) {
  union { float f; unsigned u; } v; v.f = f;
  unsigned r = v.u + 0x7FFFu + ((v.u >> 16) & 1u);   // round-to-nearest-even
  union { unsigned short s; bf16_t b; } o; o.s = (unsigned short)(r >> 16);
  return o.b;
}

// ISA 10.2: for a generic pointer into LDS, addr[31:0] is the LDS byte offset.
__device__ __forceinline__ unsigned lds_byte_off(const void* p) {
  return (unsigned)(unsigned long long)p;
}

// ---------------------------------------------------------------------------
// WMMA GEMM: out[M x 128] = epi( A[M x K] @ W[K x 128] )
//   Wt: bf16, transposed K-major: Wt[n*K + k] = W[k*128 + n]
//   CAT3: A row r = concat(H[gi[r]], H[gj[r]], A[r])  (K = 384)
//   epilogue: v = acc (+bias) (silu) (*mulv[row]) (+addv[row])
// Block = 256 threads = 8 waves; block tile 128x128; wave tile 16x128.
// ---------------------------------------------------------------------------
template <int KDIM, bool CAT3>
__global__ __launch_bounds__(256) void gemm_wmma(
    const float* __restrict__ A, const float* __restrict__ H,
    const int* __restrict__ gi, const int* __restrict__ gj,
    const bf16_t* __restrict__ Wt, const float* __restrict__ bias,
    const float* __restrict__ mulv, const float* __restrict__ addv,
    float* __restrict__ out, int M, int do_silu)
{
  __shared__ __align__(16) bf16_t Alds[128 * TK];   // [row][k]
  __shared__ __align__(16) bf16_t Wlds[128 * TK];   // [n][k]

  const int tid  = threadIdx.x;
  const int wave = tid >> 5;
  const int lane = tid & 31;
  const int half = lane >> 4;
  const int l15  = lane & 15;
  const int row0 = blockIdx.x * 128;

  f32x8 acc[8];
#pragma unroll
  for (int i = 0; i < 8; ++i)
#pragma unroll
    for (int j = 0; j < 8; ++j) acc[i][j] = 0.0f;

  for (int k0 = 0; k0 < KDIM; k0 += TK) {
    // ---- stage A tile: 128 rows x 32 k; 16 elems per thread (f32->bf16) --
    {
      const int r  = tid >> 1;
      const int ks = (tid & 1) * 16;
      const int gr = row0 + r;
      bf16_t* dst = &Alds[r * TK + ks];
      if (gr < M) {
#pragma unroll
        for (int q = 0; q < 16; ++q) {
          const int kk = k0 + ks + q;
          float v;
          if (CAT3) {
            const int sel = kk >> 7, within = kk & 127;
            if (sel == 0)      v = H[(size_t)gi[gr] * DDIM + within];
            else if (sel == 1) v = H[(size_t)gj[gr] * DDIM + within];
            else               v = A[(size_t)gr * DDIM + within];
          } else {
            v = A[(size_t)gr * KDIM + kk];
          }
          dst[q] = to_bf16(v);
        }
        if (!CAT3 && (k0 + TK < KDIM))
          __builtin_prefetch(&A[(size_t)gr * KDIM + k0 + TK], 0, 0);
      } else {
#pragma unroll
        for (int q = 0; q < 16; ++q) dst[q] = to_bf16(0.0f);
      }
    }

    // ---- stage W tile (bf16, pure copy): Wlds[n][0..31] = Wt[n*K + k0..] --
#if USE_TDM
    if (tid < 32) {
      // Tensor DMA descriptor (ISA 8.3/8.4): 2D tile 32 x 128 of 2-byte elems.
      const unsigned long long ga = (unsigned long long)(const void*)(Wt + k0);
      u32x4 g0;
      g0[0] = 1u;                                   // count=1, user mode
      g0[1] = lds_byte_off(Wlds);                   // lds_addr (bytes)
      g0[2] = (unsigned)ga;                         // global_addr[31:0]
      g0[3] = ((unsigned)(ga >> 32) & 0x01FFFFFFu)  // global_addr[56:32]
              | (2u << 30);                         // type=2 ("image")
      i32x8 g1;
      g1[0] = (int)(1u << 16);                      // wg_mask=0, data_size=1 (2B)
      g1[1] = (int)(((unsigned)KDIM & 0xFFFFu) << 16);      // tensor_dim0[15:0]
      g1[2] = (int)((((unsigned)KDIM) >> 16) | (128u << 16)); // dim0 hi | dim1 lo
      g1[3] = (int)(32u << 16);                     // dim1 hi=0 | tile_dim0=32
      g1[4] = (int)128;                             // tile_dim1=128, tile_dim2=0
      g1[5] = (int)KDIM;                            // tensor_dim0_stride[31:0]
      g1[6] = 0;                                    // stride0 hi | stride1 lo
      g1[7] = 0;
      i32x4 z4; z4[0] = 0; z4[1] = 0; z4[2] = 0; z4[3] = 0;
#if __clang_major__ >= 23
      i32x8 z8;
#pragma unroll
      for (int q = 0; q < 8; ++q) z8[q] = 0;
      __builtin_amdgcn_tensor_load_to_lds(g0, g1, z4, z4, z8, 0);
#else
      __builtin_amdgcn_tensor_load_to_lds(g0, g1, z4, z4, 0);
#endif
      __builtin_amdgcn_s_wait_tensorcnt(0);
    }
#else
#pragma unroll
    for (int s = 0; s < 2; ++s) {
      const int idx = tid + s * 256;                // 0..511 : 16B segments
      const int n = idx >> 2, ko = (idx & 3) * 8;
      *(uint4*)&Wlds[n * TK + ko] =
          *(const uint4*)&Wt[(size_t)n * KDIM + k0 + ko];
    }
#endif
    __syncthreads();

    // ---- A fragment (16-bit A 16x32 layout, ISA 7.12.2) -----------------
    bf16x16 afrag;
    const bf16_t* ap = &Alds[(wave * 16 + l15) * TK];
#pragma unroll
    for (int q = 0; q < 8; ++q) afrag[q]     = ap[half * 8 + q];        // K = 8h..
#pragma unroll
    for (int q = 0; q < 8; ++q) afrag[8 + q] = ap[16 + half * 8 + q];   // K = 16+8h..

    // ---- 8 N-tiles per wave --------------------------------------------
#pragma unroll
    for (int nt = 0; nt < 8; ++nt) {
      bf16x16 bfrag;
      const bf16_t* bp = &Wlds[(nt * 16 + l15) * TK + half * 8];        // K = 8h.., 16+8h..
#pragma unroll
      for (int q = 0; q < 16; ++q) bfrag[q] = bp[q < 8 ? q : q + 8];
      acc[nt] = __builtin_amdgcn_wmma_f32_16x16x32_bf16(
          false, afrag, false, bfrag, (short)0, acc[nt], false, false);
    }
    __syncthreads();
  }

  // ---- epilogue: D layout VGPR r -> M = r + 8*half, N = l15 -------------
#pragma unroll
  for (int nt = 0; nt < 8; ++nt) {
    const int col = nt * 16 + l15;
#pragma unroll
    for (int r = 0; r < 8; ++r) {
      const int row = row0 + wave * 16 + half * 8 + r;
      if (row < M) {
        float v = acc[nt][r];
        if (bias)    v += bias[col];
        if (do_silu) v = silu_f(v);
        const size_t off = (size_t)row * DDIM + col;
        if (mulv) v *= mulv[off];
        if (addv) v += addv[off];
        out[off] = v;
      }
    }
  }
}

// ---------------------------------------------------------------------------
// Weight pre-conversion: Wt[n*K + k] = bf16(W[k*128 + n])
// ---------------------------------------------------------------------------
__global__ void k_wconv(const float* __restrict__ W, bf16_t* __restrict__ Wt, int K) {
  long i = (long)blockIdx.x * blockDim.x + threadIdx.x;
  if (i >= (long)K * DDIM) return;
  int n = (int)(i / K), k = (int)(i % K);
  Wt[i] = to_bf16(W[(size_t)k * DDIM + n]);
}

// ---------------------------------------------------------------------------
// Small elementwise / basis / scatter kernels
// ---------------------------------------------------------------------------
__global__ void k_fill(float* p, float v, long n) {
  long i = (long)blockIdx.x * blockDim.x + threadIdx.x;
  if (i < n) p[i] = v;
}
__global__ void k_copy(float* dst, const float* src, long n) {
  long i = (long)blockIdx.x * blockDim.x + threadIdx.x;
  if (i < n) dst[i] = src[i];
}
__global__ void k_emb(const float* emb, const int* x, float* h, int N) {
  long i = (long)blockIdx.x * blockDim.x + threadIdx.x;
  if (i < (long)N * DDIM) h[i] = emb[(size_t)x[i >> 7] * DDIM + (i & 127)];
}
__global__ void k_dist(const float* pos, const int* ja, const int* ia, float* dist, int E) {
  int e = blockIdx.x * blockDim.x + threadIdx.x;
  if (e >= E) return;
  const float* a = &pos[(size_t)ia[e] * 3];
  const float* b = &pos[(size_t)ja[e] * 3];
  float dx = a[0] - b[0], dy = a[1] - b[1], dz = a[2] - b[2];
  dist[e] = sqrtf(dx * dx + dy * dy + dz * dz);
}
__device__ __forceinline__ float env_poly(float x) {
  float x5 = x * x * x * x * x;
  return 1.0f / x + (-28.0f) * x5 + 48.0f * x5 * x + (-21.0f) * x5 * x * x;
}
__global__ void k_rbf16(const float* dist, float* out, int E, float inv_cutoff) {
  int e = blockIdx.x * blockDim.x + threadIdx.x;
  if (e >= E) return;
  float x = dist[e] * inv_cutoff;
  float env = env_poly(x);
#pragma unroll
  for (int k = 0; k < 16; ++k)
    out[(size_t)e * 16 + k] = env * sinf(3.14159265358979f * (float)(k + 1) * x);
}
__global__ void k_small_mlp(const float* __restrict__ X, const float* __restrict__ W,
                            const float* __restrict__ b, float* __restrict__ out,
                            int M, int K) {
  long i = (long)blockIdx.x * blockDim.x + threadIdx.x;
  if (i >= (long)M * DDIM) return;
  int row = i >> 7, col = i & 127;
  float acc = b[col];
  for (int k = 0; k < K; ++k) acc += X[(size_t)row * K + k] * W[(size_t)k * DDIM + col];
  out[i] = silu_f(acc);
}
__global__ void k_angle(const float* pos, const int* ia, const int* ib, const int* ic,
                        float* ang, int T) {
  int t = blockIdx.x * blockDim.x + threadIdx.x;
  if (t >= T) return;
  const float* pa = &pos[(size_t)ia[t] * 3];
  const float* pb = &pos[(size_t)ib[t] * 3];
  const float* pc = &pos[(size_t)ic[t] * 3];
  float ux = pb[0] - pa[0], uy = pb[1] - pa[1], uz = pb[2] - pa[2];
  float vx = pc[0] - pb[0], vy = pc[1] - pb[1], vz = pc[2] - pb[2];
  float cx = uy * vz - uz * vy, cy = uz * vx - ux * vz, cz = ux * vy - uy * vx;
  float cr = sqrtf(cx * cx + cy * cy + cz * cz);
  float dt = ux * vx + uy * vy + uz * vz;
  ang[t] = atan2f(cr, dt);
}
__constant__ float c_Z[7][6] = {
  {3.141593f, 6.283185f, 9.424778f, 12.566371f, 15.707963f, 18.849556f},
  {4.493409f, 7.725252f, 10.904122f, 14.066194f, 17.220755f, 20.371303f},
  {5.763459f, 9.095011f, 12.322941f, 15.514603f, 18.689036f, 21.853874f},
  {6.987932f, 10.417119f, 13.698023f, 16.923621f, 20.121806f, 23.304247f},
  {8.182561f, 11.704907f, 15.039665f, 18.301256f, 21.525418f, 24.727566f},
  {9.355812f, 12.966530f, 16.354710f, 19.653152f, 22.904551f, 26.127750f},
  {10.512835f, 14.207392f, 17.647975f, 20.983463f, 24.262768f, 27.507868f}};
__device__ __forceinline__ float bessel_jl(float u, int l) {
  float j0 = sinf(u) / u;
  if (l == 0) return j0;
  float jm = j0, jc = sinf(u) / (u * u) - cosf(u) / u;
  for (int n = 1; n < l; ++n) { float t = (float)(2 * n + 1) / u * jc - jm; jm = jc; jc = t; }
  return jc;
}
__global__ void k_rad42(const float* dist, float* rad, int E) {
  int e = blockIdx.x * blockDim.x + threadIdx.x;
  if (e >= E) return;
  float x = dist[e] * 0.2f;
  float env = env_poly(x);
  for (int l = 0; l < 7; ++l)
    for (int k = 0; k < 6; ++k) {
      float z = c_Z[l][k];
      float norm = 1.41421356237f / fabsf(bessel_jl(z, l + 1));
      rad[(size_t)e * 42 + l * 6 + k] = norm * bessel_jl(x * z, l) * env;
    }
}
__global__ void k_sbf42(const float* rad, const float* ang, const int* idx_edge,
                        float* out, int T) {
  int t = blockIdx.x * blockDim.x + threadIdx.x;
  if (t >= T) return;
  float c = cosf(ang[t]);
  float pl[7];
  pl[0] = 1.0f; pl[1] = c;
  for (int l = 1; l < 6; ++l)
    pl[l + 1] = ((float)(2 * l + 1) * c * pl[l] - (float)l * pl[l - 1]) / (float)(l + 1);
  const float* r = &rad[(size_t)idx_edge[t] * 42];
  for (int l = 0; l < 7; ++l) {
    float yc = sqrtf((float)(2 * l + 1) * 0.079577471545948f) * pl[l];
    for (int k = 0; k < 6; ++k) out[(size_t)t * 42 + l * 6 + k] = r[l * 6 + k] * yc;
  }
}
__global__ void k_scatter_add(const int* seg, const float* in, float* out, int M) {
  long i = (long)blockIdx.x * blockDim.x + threadIdx.x;
  if (i >= (long)M * DDIM) return;
  int r = i >> 7, d = i & 127;
  atomicAdd(&out[(size_t)seg[r] * DDIM + d], in[i]);
}
__global__ void k_trip_scatter(const int* kj, const int* ji, const float* a,
                               const float* s, float* out, int T) {
  long i = (long)blockIdx.x * blockDim.x + threadIdx.x;
  if (i >= (long)T * DDIM) return;
  int t = i >> 7, d = i & 127;
  atomicAdd(&out[(size_t)ji[t] * DDIM + d], a[(size_t)kj[t] * DDIM + d] * s[i]);
}
__global__ void k_yfinal(const float* yt, const float* w, const float* b,
                         float* nodesum, int N) {
  int n = blockIdx.x * blockDim.x + threadIdx.x;
  if (n >= N) return;
  float acc = b[0];
  for (int d = 0; d < DDIM; ++d) acc += yt[(size_t)n * DDIM + d] * w[d];
  nodesum[n] += acc;
}
__global__ void k_batch_sum(const float* nodesum, const int* batch, float* out, int N) {
  int n = blockIdx.x * blockDim.x + threadIdx.x;
  if (n >= N) return;
  atomicAdd(&out[batch[n]], nodesum[n]);
}

// ---------------------------------------------------------------------------
// Host orchestration
// ---------------------------------------------------------------------------
static inline int cdiv(long a, long b) { return (int)((a + b - 1) / b); }

extern "C" void kernel_launch(void* const* d_in, const int* in_sizes, int n_in,
                              void* d_out, int out_size, void* d_ws, size_t ws_size,
                              hipStream_t stream) {
  // Param leaves precede the 15 array inputs (insertion order of setup_inputs()).
  const int PB = n_in - 15;
  const float* pos   = (const float*)d_in[PB + 0];
  const int* xatom   = (const int*)d_in[PB + 1];
  const int* batch   = (const int*)d_in[PB + 2];
  const int* eil     = (const int*)d_in[PB + 3];
  const int* eig     = (const int*)d_in[PB + 4];
  const int* idx_i1  = (const int*)d_in[PB + 5];
  const int* idx_j   = (const int*)d_in[PB + 6];
  const int* idx_k   = (const int*)d_in[PB + 7];
  const int* idx_kj  = (const int*)d_in[PB + 8];
  const int* idx_ji1 = (const int*)d_in[PB + 9];
  const int* idx_i2  = (const int*)d_in[PB + 10];
  const int* idx_j1  = (const int*)d_in[PB + 11];
  const int* idx_j2  = (const int*)d_in[PB + 12];
  const int* idx_jj  = (const int*)d_in[PB + 13];
  const int* idx_ji2 = (const int*)d_in[PB + 14];

  const int N  = in_sizes[PB + 1];
  const int El = in_sizes[PB + 3] / 2;
  const int Eg = in_sizes[PB + 4] / 2;
  const int T1 = in_sizes[PB + 5];
  const int T2 = in_sizes[PB + 10];

  const int* jl_ = eil;           // edge_index_l[0] (source j)
  const int* il_ = eil + El;      // edge_index_l[1] (target i)
  const int* jg_ = eig;
  const int* ig_ = eig + Eg;

  // Param flatten (insertion order):
  // 0 emb | 1,2 rbf_g_mlp | 3,4 rbf_l_mlp | 5,6 sbf1_mlp | 7,8 sbf2_mlp
  // global L: base 9+19L : hmlp(0,1) res1(2..5) res2(6..9) res3(10..13)
  //                        mlp(14,15) xedge(16,17) lin(18)
  // local  L: base 123+41L: hmlp(0,1) kj(2,3) ji1(4,5) ji2(6,7) jj(8,9)
  //   sbf1a(10,11) sbf1b(12,13) sbf2a(14,15) sbf2b(16,17) rbf1(18) rbf2(19)
  //   rbfout(20) res1(21..24) res2(25..28) res3(29..32) y1(33,34) y2(35,36)
  //   y3(37,38) yW.w(39) yW.b(40)
  auto P  = [&](int i) -> const float* { return (const float*)d_in[i]; };
  auto GP = [&](int L, int o) -> const float* { return P(9 + 19 * L + o); };
  auto LP = [&](int L, int o) -> const float* { return P(9 + 19 * 6 + 41 * L + o); };

  // ---- workspace bump allocator ----
  char* wsp = (char*)d_ws;
  size_t used = 0;
  auto alloc = [&](size_t nf) -> float* {
    float* p = (float*)(wsp + used);
    used += ((nf * sizeof(float) + 255) / 256) * 256;
    return p;
  };
  const int Tm = (T1 > T2 ? T1 : T2);
  const int Em = (Eg > El ? Eg : El);
  float* h     = alloc((size_t)N * DDIM);
  float* hres  = alloc((size_t)N * DDIM);
  float* ht1   = alloc((size_t)N * DDIM);
  float* ht2   = alloc((size_t)N * DDIM);
  float* nsum  = alloc((size_t)N);
  float* dl    = alloc((size_t)El);
  float* dg    = alloc((size_t)Eg);
  float* r16   = alloc((size_t)Em * 16);
  float* rbfg  = alloc((size_t)Eg * DDIM);
  float* rbfl  = alloc((size_t)El * DDIM);
  float* xe    = alloc((size_t)Eg * DDIM);
  float* msg   = alloc((size_t)Eg * DDIM);
  float* mbuf  = alloc((size_t)El * DDIM);
  float* mk    = alloc((size_t)El * DDIM);
  float* macc  = alloc((size_t)El * DDIM);
  float* sbf1  = alloc((size_t)T1 * DDIM);
  float* sbf2  = alloc((size_t)T2 * DDIM);
  float* tA    = alloc((size_t)Tm * DDIM);
  float* tB    = alloc((size_t)Tm * DDIM);
  float* rad   = alloc((size_t)El * 42);
  float* sph   = alloc((size_t)Tm * 42);
  float* ang1  = alloc((size_t)T1);
  float* ang2  = alloc((size_t)T2);
  bf16_t* wt   = (bf16_t*)alloc(384 * DDIM / 2 + 64);   // bf16 weight scratch
  (void)ws_size;

  auto fill = [&](float* p, float v, long n) {
    if (n > 0) k_fill<<<cdiv(n, 256), 256, 0, stream>>>(p, v, n);
  };
  auto g128 = [&](const float* A, const float* W, const float* b, const float* mul,
                  const float* add, float* o, int M, int act) {
    if (M <= 0) return;
    k_wconv<<<cdiv((long)128 * DDIM, 256), 256, 0, stream>>>(W, wt, 128);
    gemm_wmma<128, false><<<cdiv(M, 128), 256, 0, stream>>>(
        A, nullptr, nullptr, nullptr, wt, b, mul, add, o, M, act);
  };
  auto g384 = [&](const float* feat, const float* H, const int* gi, const int* gj,
                  const float* W, const float* b, const float* mul, const float* add,
                  float* o, int M, int act) {
    if (M <= 0) return;
    k_wconv<<<cdiv((long)384 * DDIM, 256), 256, 0, stream>>>(W, wt, 384);
    gemm_wmma<384, true><<<cdiv(M, 128), 256, 0, stream>>>(
        feat, H, gi, gj, wt, b, mul, add, o, M, act);
  };

  // ---- precompute ----
  k_emb<<<cdiv((long)N * DDIM, 256), 256, 0, stream>>>(P(0), xatom, h, N);
  if (Eg > 0) k_dist<<<cdiv(Eg, 256), 256, 0, stream>>>(pos, jg_, ig_, dg, Eg);
  if (El > 0) k_dist<<<cdiv(El, 256), 256, 0, stream>>>(pos, jl_, il_, dl, El);
  if (Eg > 0) {
    k_rbf16<<<cdiv(Eg, 256), 256, 0, stream>>>(dg, r16, Eg, 0.2f);
    k_small_mlp<<<cdiv((long)Eg * DDIM, 256), 256, 0, stream>>>(r16, P(1), P(2), rbfg, Eg, 16);
  }
  if (El > 0) {
    k_rbf16<<<cdiv(El, 256), 256, 0, stream>>>(dl, r16, El, 0.2f);
    k_small_mlp<<<cdiv((long)El * DDIM, 256), 256, 0, stream>>>(r16, P(3), P(4), rbfl, El, 16);
    k_rad42<<<cdiv(El, 128), 128, 0, stream>>>(dl, rad, El);
  }
  if (T1 > 0) {
    k_angle<<<cdiv(T1, 256), 256, 0, stream>>>(pos, idx_i1, idx_j, idx_k, ang1, T1);
    k_sbf42<<<cdiv(T1, 128), 128, 0, stream>>>(rad, ang1, idx_kj, sph, T1);
    k_small_mlp<<<cdiv((long)T1 * DDIM, 256), 256, 0, stream>>>(sph, P(5), P(6), sbf1, T1, 42);
  }
  if (T2 > 0) {
    k_angle<<<cdiv(T2, 256), 256, 0, stream>>>(pos, idx_i2, idx_j1, idx_j2, ang2, T2);
    k_sbf42<<<cdiv(T2, 128), 128, 0, stream>>>(rad, ang2, idx_jj, sph, T2);
    k_small_mlp<<<cdiv((long)T2 * DDIM, 256), 256, 0, stream>>>(sph, P(7), P(8), sbf2, T2, 42);
  }
  fill(nsum, 0.0f, N);
  fill((float*)d_out, 0.0f, out_size);

  for (int L = 0; L < 6; ++L) {
    // ================= global_mp =================
    k_copy<<<cdiv((long)N * DDIM, 256), 256, 0, stream>>>(hres, h, (long)N * DDIM);
    g128(h, GP(L, 0), GP(L, 1), nullptr, nullptr, h, N, 1);                 // h = mlp(h,hmlp)
    g384(rbfg, h, ig_, jg_, GP(L, 16), GP(L, 17), nullptr, nullptr, xe, Eg, 1);  // x_edge
    g128(rbfg, GP(L, 18), nullptr, xe, nullptr, msg, Eg, 0);                // msg=(ea@lin)*xe
    if (Eg > 0)
      k_scatter_add<<<cdiv((long)Eg * DDIM, 256), 256, 0, stream>>>(ig_, msg, h, Eg);
    // res1
    g128(h, GP(L, 2), GP(L, 3), nullptr, nullptr, ht1, N, 1);
    g128(ht1, GP(L, 4), GP(L, 5), nullptr, h, h, N, 1);
    // h = mlp(h,'mlp') + res_h
    g128(h, GP(L, 14), GP(L, 15), nullptr, hres, h, N, 1);
    // res2, res3
    g128(h, GP(L, 6), GP(L, 7), nullptr, nullptr, ht1, N, 1);
    g128(ht1, GP(L, 8), GP(L, 9), nullptr, h, h, N, 1);
    g128(h, GP(L, 10), GP(L, 11), nullptr, nullptr, ht1, N, 1);
    g128(ht1, GP(L, 12), GP(L, 13), nullptr, h, h, N, 1);

    // ================= local_mp =================
    k_copy<<<cdiv((long)N * DDIM, 256), 256, 0, stream>>>(hres, h, (long)N * DDIM);
    g128(h, LP(L, 0), LP(L, 1), nullptr, nullptr, h, N, 1);                 // h = mlp(h,hmlp)
    // m_kj = mlp(m,kj) * (rbf@rbf1)
    g384(rbfl, h, il_, jl_, LP(L, 2), LP(L, 3), nullptr, nullptr, mk, El, 1);
    g128(rbfl, LP(L, 18), nullptr, mk, nullptr, mk, El, 0);
    // sbf1 path + triplet scatter
    g128(sbf1, LP(L, 10), LP(L, 11), nullptr, nullptr, tA, T1, 1);
    g128(tA, LP(L, 12), LP(L, 13), nullptr, nullptr, tB, T1, 1);
    fill(macc, 0.0f, (long)El * DDIM);
    if (T1 > 0)
      k_trip_scatter<<<cdiv((long)T1 * DDIM, 256), 256, 0, stream>>>(idx_kj, idx_ji1, mk, tB, macc, T1);
    g384(rbfl, h, il_, jl_, LP(L, 4), LP(L, 5), nullptr, macc, mbuf, El, 1);  // m = mlp(m,ji1)+m_kj
    // m_jj = mlp(m,jj) * (rbf@rbf2)
    g128(mbuf, LP(L, 8), LP(L, 9), nullptr, nullptr, mk, El, 1);
    g128(rbfl, LP(L, 19), nullptr, mk, nullptr, mk, El, 0);
    g128(sbf2, LP(L, 14), LP(L, 15), nullptr, nullptr, tA, T2, 1);
    g128(tA, LP(L, 16), LP(L, 17), nullptr, nullptr, tB, T2, 1);
    fill(macc, 0.0f, (long)El * DDIM);
    if (T2 > 0)
      k_trip_scatter<<<cdiv((long)T2 * DDIM, 256), 256, 0, stream>>>(idx_jj, idx_ji2, mk, tB, macc, T2);
    g128(mbuf, LP(L, 6), LP(L, 7), nullptr, macc, mbuf, El, 1);             // m = mlp(m,ji2)+m_jj
    g128(rbfl, LP(L, 20), nullptr, mbuf, nullptr, mbuf, El, 0);             // m = (rbf@rbfout)*m
    // h = segment_sum(m, i)
    fill(h, 0.0f, (long)N * DDIM);
    if (El > 0)
      k_scatter_add<<<cdiv((long)El * DDIM, 256), 256, 0, stream>>>(il_, mbuf, h, El);
    // res1
    g128(h, LP(L, 21), LP(L, 22), nullptr, nullptr, ht1, N, 1);
    g128(ht1, LP(L, 23), LP(L, 24), nullptr, h, h, N, 1);
    // h = mlp(h,hmlp) + res_h (shared hmlp)
    g128(h, LP(L, 0), LP(L, 1), nullptr, hres, h, N, 1);
    // res2, res3
    g128(h, LP(L, 25), LP(L, 26), nullptr, nullptr, ht1, N, 1);
    g128(ht1, LP(L, 27), LP(L, 28), nullptr, h, h, N, 1);
    g128(h, LP(L, 29), LP(L, 30), nullptr, nullptr, ht1, N, 1);
    g128(ht1, LP(L, 31), LP(L, 32), nullptr, h, h, N, 1);
    // y head
    g128(h, LP(L, 33), LP(L, 34), nullptr, nullptr, ht1, N, 1);
    g128(ht1, LP(L, 35), LP(L, 36), nullptr, nullptr, ht2, N, 1);
    g128(ht2, LP(L, 37), LP(L, 38), nullptr, nullptr, ht1, N, 1);
    k_yfinal<<<cdiv(N, 256), 256, 0, stream>>>(ht1, LP(L, 39), LP(L, 40), nsum, N);
  }

  k_batch_sum<<<cdiv(N, 256), 256, 0, stream>>>(nsum, batch, (float*)d_out, N);
}
